// LIIF_1803886264344
// MI455X (gfx1250) — matrix operand order repaired and checked by reading, not compiled, not run on hardware
//
#include <hip/hip_runtime.h>
#include <hip/hip_bf16.h>

// ---------------------------------------------------------------------------
// LIIF feature-sampling forward for MI455X (gfx1250).
//
// Roofline (MI455X: 23.3 TB/s HBM, 192 MB L2, wave32 WMMA):
//   - conv encoder: 113 MFLOP -> one v_wmma_f32_16x16x32_f16 per 16x16 tile
//     (K=27 padded to 32), 8192 WMMAs total: noise.
//   - cont output: 8*4096*4*580*4B = 304 MB coalesced stores -> ~13 us. This
//     is the floor; everything else must stay under it.
//   - gather reads: logical 302 MB but physical footprint is the 8.5 MB
//     padded feature map (L2-resident, ~36x reuse) -> cache-served.
// v2 changes vs v1 (which was VALU-bound in the gather, ~15 ops/channel):
//   - zero-padded NHWC feature map [B,66,66,64] removes unfold bounds checks
//   - 576-entry LDS offset table removes per-channel /9, /3, address math:
//     inner loop = ds_load_b128 + 4 gathers + global_store_b128
//   - float4 output stores (4x b128 + 1x b64 per corner instead of 18x b32)
// ---------------------------------------------------------------------------

typedef __attribute__((ext_vector_type(16))) _Float16 v16h;
typedef __attribute__((ext_vector_type(8)))  float    v8f;

#define BB_    8
#define HH_    64
#define WW_    64
#define CIN_   3
#define COUT_  64
#define QQ_    4096
#define KTOT_  27                        // C_IN*3*3
#define HP_    66                        // padded spatial dim
#define FEATP_ELEMS (BB_ * HP_ * HP_ * COUT_)   // 2,230,272 floats (8.5 MB)
#define CONT_ELEMS  76021760ull          // 8*4096*4*580
#define AREA_ELEMS  131072ull            // 8*4096*4
#define EPS_   1e-6f

// -------------------------------------------------------------------------
// Kernel 0: zero the padded feature map (workspace is not re-zeroed by the
// harness; border must be 0 each call = unfold's zero padding).
// 2,230,272 / 4 = 557,568 float4 = 2178 blocks * 256 threads exactly.
// -------------------------------------------------------------------------
__global__ void liif_zero(float* __restrict__ p)
{
    const size_t i = ((size_t)blockIdx.x * 256 + threadIdx.x) * 4;
    *(float4*)(p + i) = make_float4(0.f, 0.f, 0.f, 0.f);
}

// -------------------------------------------------------------------------
// Kernel 1: encoder conv as implicit GEMM with WMMA.
//   feat[c, n] = sum_k W[c,k] * im2col[k,n] + bias[c],  n = (b,y,x) flat.
//   One wave owns one 16-wide N tile and all 4 M tiles (reuses B operand).
//   Output stored zero-padded channels-last: feat[b, y+1, x+1, c].
// -------------------------------------------------------------------------
__global__ void liif_conv_wmma(const float* __restrict__ lr,
                               const float* __restrict__ wts,   // [64,27]
                               const float* __restrict__ bias,  // [64]
                               float* __restrict__ feat)        // [B,66,66,64]
{
    const int lane  = threadIdx.x & 31;
    const int wave  = threadIdx.x >> 5;
    const int ntile = blockIdx.x * 8 + wave;    // 2048 N tiles of 16
    const int nloc  = lane & 15;
    const int khalf = lane >> 4;                // 0: lanes 0-15, 1: lanes 16-31
    const int n     = ntile * 16 + nloc;        // flat (b,y,x)
    const int bb    = n >> 12;
    const int rem   = n & 4095;
    const int y     = rem >> 6;
    const int x     = rem & 63;

    // ---- B tile: im2col, 32x16 f16 (K padded 27->32 with zeros).
    // B k order used here (contiguous, same as A): lanes 0-15 hold K=0..15, lanes 16-31 K=16..31;
    // VGPR v packs K = 16*khalf + 2v (+1).
    v16h bmat;
#pragma unroll
    for (int v = 0; v < 8; ++v) {
#pragma unroll
        for (int h = 0; h < 2; ++h) {
            const int k = khalf * 16 + 2 * v + h;
            float val = 0.0f;
            if (k < KTOT_) {
                const int cin = k / 9;
                const int r   = k - 9 * cin;
                const int kh  = r / 3;
                const int kw  = r - 3 * kh;
                const int y2  = y + kh - 1;
                const int x2  = x + kw - 1;
                if ((unsigned)y2 < 64u && (unsigned)x2 < 64u)
                    val = lr[((bb * CIN_ + cin) * HH_ + y2) * WW_ + x2];
            }
            bmat[2 * v + h] = (_Float16)val;
        }
    }

    const int m = lane & 15;
    // padded NHWC destination for this lane's position
    float* drow = feat + (((size_t)bb * HP_ + (y + 1)) * HP_ + (x + 1)) * COUT_;

#pragma unroll
    for (int mt = 0; mt < 4; ++mt) {
        // ---- A tile: weights, 16x32 f16. kbase = 2v + 16*khalf.
        v16h amat;
#pragma unroll
        for (int v = 0; v < 8; ++v) {
            const int kbase = 2 * v + 16 * khalf;
#pragma unroll
            for (int h = 0; h < 2; ++h) {
                const int k = kbase + h;
                const float val = (k < KTOT_) ? wts[(mt * 16 + m) * KTOT_ + k] : 0.0f;
                amat[2 * v + h] = (_Float16)val;
            }
        }

        // ---- C/D layout: VGPR v holds M = v + 8*khalf, N = lane&15.
        v8f acc;
#pragma unroll
        for (int v = 0; v < 8; ++v)
            acc[v] = bias[mt * 16 + v + 8 * khalf];

        acc = __builtin_amdgcn_wmma_f32_16x16x32_f16(
                  /*neg_a=*/false, amat, /*neg_b=*/false, bmat,
                  /*c_mod=*/(short)0, acc, /*reuse_a=*/false, /*reuse_b=*/false);

        // channels mt*16 + 8*khalf + {0..7} contiguous -> 2x float4 store.
        float* dst = drow + mt * 16 + 8 * khalf;
        *(float4*)(dst + 0) = make_float4(acc[0], acc[1], acc[2], acc[3]);
        *(float4*)(dst + 4) = make_float4(acc[4], acc[5], acc[6], acc[7]);
    }
}

// -------------------------------------------------------------------------
// grid_sample(nearest, reflection, align_corners=False) index for size 64.
// jnp.round is round-half-even -> rintf under default RNE.
// -------------------------------------------------------------------------
__device__ __forceinline__ int liif_to_index(float coord)
{
    float ixf   = ((coord + 1.0f) * 64.0f - 1.0f) * 0.5f;
    float xa    = fabsf(ixf + 0.5f);            // reflect over [-0.5, 63.5]
    float ext   = fmodf(xa, 64.0f);
    float flips = floorf(xa * (1.0f / 64.0f));
    float r     = (fmodf(flips, 2.0f) == 0.0f) ? (-0.5f + ext) : (63.5f - ext);
    r = fminf(fmaxf(r, 0.0f), 63.0f);
    return (int)rintf(r);
}

// -------------------------------------------------------------------------
// Kernel 2: per-(b,q,corner) gather + assemble. One wave per corner.
// LDS table tab[ch] = (kh*66 + kw)*64 + c  with ch = c*9 + kh*3 + kw gives
// the gather offset relative to the wave-uniform base (iy, ix) in the padded
// map; no bounds checks needed. Stores are float4-coalesced (lane == ch/4).
// -------------------------------------------------------------------------
__global__ void liif_gather(const float* __restrict__ feat,     // [B,66,66,64]
                            const float* __restrict__ h_coord,  // [B,Q,2]
                            const float* __restrict__ cell,     // [B,Q,2]
                            float* __restrict__ out)
{
    __shared__ unsigned tab[576];
    for (int t = threadIdx.x; t < 576; t += 256) {
        const int c  = t / 9;
        const int r  = t - 9 * c;
        const int kh = r / 3;
        const int kw = r - 3 * kh;
        tab[t] = (unsigned)((kh * HP_ + kw) * COUT_ + c);
    }
    __syncthreads();

    const int lane = threadIdx.x & 31;
    const int gid  = blockIdx.x * 8 + (threadIdx.x >> 5);  // corner id < 131072
    const int cr   = gid & 3;
    const int q    = (gid >> 2) & (QQ_ - 1);
    const int bb   = gid >> 14;

    const float hc_h = h_coord[(bb * QQ_ + q) * 2 + 0];
    const float hc_w = h_coord[(bb * QQ_ + q) * 2 + 1];

    const float chf = 1.0f / 64.0f;
    const float oh  = (cr < 2)        ? (-chf + EPS_) : (chf - EPS_);
    const float ow  = ((cr & 1) == 0) ? (-chf + EPS_) : (chf - EPS_);

    const int iy = liif_to_index(hc_h + oh);
    const int ix = liif_to_index(hc_w + ow);

    const float lat_h = -1.0f + (2.0f * (float)iy + 1.0f) * (1.0f / 64.0f);
    const float lat_w = -1.0f + (2.0f * (float)ix + 1.0f) * (1.0f / 64.0f);
    const float rel_h = (hc_h - lat_h) * 64.0f;
    const float rel_w = (hc_w - lat_w) * 64.0f;

    const size_t obase = (size_t)gid * 580;         // cont[b,q,cr,:]
    // Base at padded (iy, ix): unpadded y2 = iy+kh-1 -> padded row iy+kh.
    const float* fbase = feat + (((size_t)bb * HP_ + iy) * HP_ + ix) * COUT_;

#pragma unroll
    for (int it = 0; it < 4; ++it) {                // 4*128 = 512 channels
        const int ch0 = it * 128 + lane * 4;
        const uint4 o = *(const uint4*)&tab[ch0];   // ds_load_b128
        float4 v;
        v.x = fbase[o.x];
        v.y = fbase[o.y];
        v.z = fbase[o.z];
        v.w = fbase[o.w];
        *(float4*)(out + obase + ch0) = v;          // coalesced b128
    }
    {                                               // tail: channels 512..575
        const int ch0 = 512 + lane * 2;
        const uint2 o = *(const uint2*)&tab[ch0];
        float2 v;
        v.x = fbase[o.x];
        v.y = fbase[o.y];
        *(float2*)(out + obase + ch0) = v;          // coalesced b64
    }

    if (lane < 4) {                                 // channels 576..579
        float v;
        if (lane == 0)      v = rel_h;
        else if (lane == 1) v = rel_w;
        else                v = cell[(bb * QQ_ + q) * 2 + (lane - 2)] * 64.0f;
        out[obase + 576 + lane] = v;
    }
    if (lane == 0)
        out[CONT_ELEMS + (size_t)gid] = fabsf(rel_h * rel_w);   // areas[b,q,cr]
    if (gid == 0 && lane == 0)
        out[CONT_ELEMS + AREA_ELEMS] = 580.0f;                  // channel scalar
}

// -------------------------------------------------------------------------
extern "C" void kernel_launch(void* const* d_in, const int* in_sizes, int n_in,
                              void* d_out, int out_size, void* d_ws, size_t ws_size,
                              hipStream_t stream)
{
    (void)in_sizes; (void)n_in; (void)out_size; (void)ws_size;
    const float* lr      = (const float*)d_in[0];  // [8,3,64,64]
    const float* wts     = (const float*)d_in[1];  // [64,3,3,3]
    const float* bias    = (const float*)d_in[2];  // [64]
    const float* h_coord = (const float*)d_in[3];  // [8,4096,2]
    const float* cell    = (const float*)d_in[4];  // [8,4096,2]
    float* out  = (float*)d_out;
    float* feat = (float*)d_ws;                    // padded NHWC, 8.5 MB

    // 557,568 float4 fills = 2178 blocks * 256 threads (exact).
    liif_zero<<<2178, 256, 0, stream>>>(feat);
    // 2048 N-tiles / 8 waves per block = 256 blocks.
    liif_conv_wmma<<<256, 256, 0, stream>>>(lr, wts, bias, feat);
    // 131072 corners, one wave each, 8 waves per block.
    liif_gather<<<16384, 256, 0, stream>>>(feat, h_coord, cell, out);
}
